// GATModelVAE_30769145708621
// MI455X (gfx1250) — matrix operand order repaired and checked by reading, not compiled, not run on hardware
//
#include <hip/hip_runtime.h>
#include <math.h>

typedef __attribute__((ext_vector_type(16))) _Float16 v16h;
typedef __attribute__((ext_vector_type(8)))  float    v8f;

#define NEG_SLOPE 0.2f

// ---------------------------------------------------------------------------
// f32 -> f16 weight conversion (done once per conv; W is tiny and reused by
// every M-tile, so convert up front instead of per k-step in the GEMM).
// ---------------------------------------------------------------------------
__global__ void cvt_f32_to_f16_kernel(const float* __restrict__ in,
                                      _Float16* __restrict__ out, int n) {
  int i = blockIdx.x * blockDim.x + threadIdx.x;
  if (i < n) out[i] = (_Float16)in[i];
}

// ---------------------------------------------------------------------------
// GEMM: Hout[M,Nc] = X[M,K] @ W16[K,Nc]  (f16 WMMA, f32 accumulate)
// Block = 8 waves sharing one N-tile: B tile (K x 16 f16) staged in LDS once,
// each wave keeps its B fragments in registers and sweeps 4 M-tiles.
// K templated (128 / 32) so the k-loop fully unrolls into WMMA chains.
// All branches around WMMA are wave-uniform -> EXEC stays all-ones (ISA req).
// ---------------------------------------------------------------------------
template <int K>
__global__ __launch_bounds__(256) void gemm_wmma_lds_kernel(
    const float* __restrict__ X, const _Float16* __restrict__ W16,
    float* __restrict__ Hout, int M, int Nc) {
  __shared__ _Float16 Bs[K * 16];

  const int tn = blockIdx.x;                       // N-tile for whole block
  // cooperative stage of B tile (K x 16) into LDS
  for (int t = threadIdx.x; t < K * 16; t += 256) {
    int k = t >> 4, n = t & 15;
    Bs[t] = W16[(size_t)k * Nc + tn * 16 + n];
  }
  __syncthreads();

  const int lane  = threadIdx.x & 31;
  const int wave  = threadIdx.x >> 5;
  const int colL  = lane & 15;
  const int tilesM = M >> 4;

  // per-wave B fragments from LDS (ds_load), reused across 4 M-tiles
  const int khalfB = (lane >> 4) << 4;             // lanes>=16 hold K+16 half
  v16h bfrag[K / 32];
#pragma unroll
  for (int kk = 0; kk < K / 32; ++kk) {
#pragma unroll
    for (int j = 0; j < 16; ++j)
      bfrag[kk][j] = Bs[(kk * 32 + khalfB + j) * 16 + colL];
  }

  const int khalfA = (lane >> 4) << 4;             // lanes>=16 hold K+16 half
#pragma unroll
  for (int i = 0; i < 4; ++i) {
    const int tm = (blockIdx.y * 4 + i) * 8 + wave;
    if (tm >= tilesM) continue;                    // wave-uniform guard
    const int rowA = tm * 16 + colL;
    v8f acc = {};
#pragma unroll
    for (int kk = 0; kk < K / 32; ++kk) {
      v16h a;
      const float* xrow = X + (size_t)rowA * K + kk * 32 + khalfA;
#pragma unroll
      for (int j = 0; j < 8; ++j) {
        a[j]     = (_Float16)xrow[j];              // K = kk*32+khalfA+j
        a[j + 8] = (_Float16)xrow[j + 8];          // K = +8
      }
      acc = __builtin_amdgcn_wmma_f32_16x16x32_f16(false, a, false, bfrag[kk],
                                                   (short)0, acc, false, false);
    }
    const int r0 = tm * 16 + ((lane >> 4) << 3);
#pragma unroll
    for (int r = 0; r < 8; ++r)
      Hout[(size_t)(r0 + r) * Nc + tn * 16 + colL] = acc[r];
  }
}

// ---------------------------------------------------------------------------
// Attention logits: al_s[n,h] = dot(h[n,h,:], a_src[h,:]), same for a_dst
// ---------------------------------------------------------------------------
__global__ void alpha_kernel(const float* __restrict__ h,
                             const float* __restrict__ a_src,
                             const float* __restrict__ a_dst,
                             float* __restrict__ al_s, float* __restrict__ al_d,
                             int Nn, int Hh, int O) {
  int i = blockIdx.x * blockDim.x + threadIdx.x;
  if (i >= Nn * Hh) return;
  int n = i / Hh, hh = i - n * Hh;
  const float* hp = h + ((size_t)n * Hh + hh) * O;
  const float* as = a_src + (size_t)hh * O;
  const float* ad = a_dst + (size_t)hh * O;
  float ss = 0.f, sd = 0.f;
  for (int o = 0; o < O; ++o) { float v = hp[o]; ss += v * as[o]; sd += v * ad[o]; }
  al_s[i] = ss; al_d[i] = sd;
}

// ---------------------------------------------------------------------------
// Init: emax=-inf, denom=0 (N*H), agg=0 (N*F)
// ---------------------------------------------------------------------------
__global__ void init_kernel(float* __restrict__ emax, float* __restrict__ denom,
                            float* __restrict__ agg, int nh, int nf) {
  int i = blockIdx.x * blockDim.x + threadIdx.x;
  if (i < nf) agg[i] = 0.f;
  if (i < nh) { emax[i] = -__builtin_huge_valf(); denom[i] = 0.f; }
}

// float atomic-max via ordered-int trick (signed max for >=0, unsigned min for <0)
__device__ __forceinline__ void atomicMaxF(float* addr, float val) {
  if (val >= 0.0f) atomicMax((int*)addr, __float_as_int(val));
  else             atomicMin((unsigned int*)addr, __float_as_uint(val));
}

__device__ __forceinline__ void edge_sd(const int* __restrict__ ei, int e, int E,
                                        int& s, int& d) {
  if (e < E) { s = ei[e]; d = ei[E + e]; }
  else       { s = e - E; d = e - E; }          // self-loops appended
}

// ---------------------------------------------------------------------------
// Pass 1: per-dst segment max of leaky_relu(al_s[src]+al_d[dst])
// ---------------------------------------------------------------------------
__global__ void edge_max_kernel(const int* __restrict__ ei, int E, int Nn,
                                const float* __restrict__ al_s,
                                const float* __restrict__ al_d,
                                float* __restrict__ emax, int Hh) {
  int e = blockIdx.x * blockDim.x + threadIdx.x;
  int EA = E + Nn;
  if (e >= EA) return;
  int s, d; edge_sd(ei, e, E, s, d);
  for (int hh = 0; hh < Hh; ++hh) {
    float v = al_s[(size_t)s * Hh + hh] + al_d[(size_t)d * Hh + hh];
    v = v > 0.f ? v : NEG_SLOPE * v;
    atomicMaxF(&emax[(size_t)d * Hh + hh], v);
  }
}

// ---------------------------------------------------------------------------
// Pass 2: w = exp(e - emax[dst]); denom[dst] += w; agg[dst,h,:] += w*h[src,h,:]
// One thread per (edge, head); float4 gathers + global_atomic_add_f32.
// Source row prefetched (global_prefetch_b8) while the exp path computes.
// ---------------------------------------------------------------------------
__global__ void edge_acc_kernel(const int* __restrict__ ei, int E, int Nn,
                                const float* __restrict__ al_s,
                                const float* __restrict__ al_d,
                                const float* __restrict__ emax,
                                const float* __restrict__ hfeat,
                                float* __restrict__ denom, float* __restrict__ agg,
                                int Hh, int O) {
  int i = blockIdx.x * blockDim.x + threadIdx.x;
  int EA = E + Nn;
  if (i >= EA * Hh) return;
  int e = i / Hh, hh = i - e * Hh;
  int s, d; edge_sd(ei, e, E, s, d);
  int F = Hh * O;
  const float* hp = hfeat + (size_t)s * F + hh * O;
  __builtin_prefetch(hp, 0, 1);                  // hide gather latency
  float v = al_s[(size_t)s * Hh + hh] + al_d[(size_t)d * Hh + hh];
  v = v > 0.f ? v : NEG_SLOPE * v;
  float m = emax[(size_t)d * Hh + hh];
  if (!(m >= -3.0e38f)) m = 0.f;                 // reference's isfinite guard
  float w = expf(v - m);
  atomicAdd(&denom[(size_t)d * Hh + hh], w);
  const float4* hp4 = (const float4*)hp;         // O is 16/32/128 -> 16B aligned
  float*        ap  = agg + (size_t)d * F + hh * O;
  for (int o = 0; o < (O >> 2); ++o) {
    float4 hv = hp4[o];
    atomicAdd(&ap[4 * o + 0], w * hv.x);
    atomicAdd(&ap[4 * o + 1], w * hv.y);
    atomicAdd(&ap[4 * o + 2], w * hv.z);
    atomicAdd(&ap[4 * o + 3], w * hv.w);
  }
}

// ---------------------------------------------------------------------------
// Finish: out = agg/denom + bias (optional ReLU)
// ---------------------------------------------------------------------------
__global__ void finish_kernel(const float* __restrict__ agg,
                              const float* __restrict__ denom,
                              const float* __restrict__ bias,
                              float* __restrict__ out,
                              int Nn, int Hh, int O, int relu) {
  int i = blockIdx.x * blockDim.x + threadIdx.x;
  int F = Hh * O;
  if (i >= Nn * F) return;
  int n = i / F, f = i - n * F;
  float den = denom[(size_t)n * Hh + f / O];
  float v = agg[i] / den + bias[f];
  if (relu) v = v > 0.f ? v : 0.f;
  out[i] = v;
}

// ---------------------------------------------------------------------------
// z = eps * exp(logvar) + mu
// ---------------------------------------------------------------------------
__global__ void z_kernel(const float* __restrict__ eps,
                         const float* __restrict__ logvar,
                         const float* __restrict__ mu,
                         float* __restrict__ z, int n) {
  int i = blockIdx.x * blockDim.x + threadIdx.x;
  if (i >= n) return;
  z[i] = eps[i] * expf(logvar[i]) + mu[i];
}

// ---------------------------------------------------------------------------
static inline int ceil_div(int a, int b) { return (a + b - 1) / b; }

static void run_gat(const float* xin, int in_dim,
                    const float* W, const float* a_s, const float* a_d,
                    const float* bias, int heads, int O,
                    float* out, int relu,
                    const int* ei, int E, int Nn,
                    float* hbuf, float* agg, float* als, float* ald,
                    float* emax, float* denom, _Float16* w16,
                    hipStream_t stream) {
  const int F  = heads * O;
  const int EA = E + Nn;
  const int tilesM = Nn / 16;

  cvt_f32_to_f16_kernel<<<ceil_div(in_dim * F, 256), 256, 0, stream>>>(W, w16, in_dim * F);
  dim3 ggrid(F / 16, ceil_div(tilesM, 32));
  if (in_dim == 128)
    gemm_wmma_lds_kernel<128><<<ggrid, 256, 0, stream>>>(xin, w16, hbuf, Nn, F);
  else
    gemm_wmma_lds_kernel<32><<<ggrid, 256, 0, stream>>>(xin, w16, hbuf, Nn, F);

  alpha_kernel<<<ceil_div(Nn * heads, 256), 256, 0, stream>>>(hbuf, a_s, a_d, als, ald, Nn, heads, O);
  init_kernel<<<ceil_div(Nn * F, 256), 256, 0, stream>>>(emax, denom, agg, Nn * heads, Nn * F);
  edge_max_kernel<<<ceil_div(EA, 256), 256, 0, stream>>>(ei, E, Nn, als, ald, emax, heads);
  edge_acc_kernel<<<ceil_div(EA * heads, 256), 256, 0, stream>>>(ei, E, Nn, als, ald, emax,
                                                                 hbuf, denom, agg, heads, O);
  finish_kernel<<<ceil_div(Nn * F, 256), 256, 0, stream>>>(agg, denom, bias, out, Nn, heads, O, relu);
}

extern "C" void kernel_launch(void* const* d_in, const int* in_sizes, int n_in,
                              void* d_out, int out_size, void* d_ws, size_t ws_size,
                              hipStream_t stream) {
  const float* x    = (const float*)d_in[0];
  const int*   ei   = (const int*)  d_in[1];
  const float* eps  = (const float*)d_in[2];
  const float* w1   = (const float*)d_in[3];
  const float* a1s  = (const float*)d_in[4];
  const float* a1d  = (const float*)d_in[5];
  const float* b1   = (const float*)d_in[6];
  const float* wmu  = (const float*)d_in[7];
  const float* amus = (const float*)d_in[8];
  const float* amud = (const float*)d_in[9];
  const float* bmu  = (const float*)d_in[10];
  const float* wlv  = (const float*)d_in[11];
  const float* alvs = (const float*)d_in[12];
  const float* alvd = (const float*)d_in[13];
  const float* blv  = (const float*)d_in[14];
  const float* wd1  = (const float*)d_in[15];
  const float* ad1s = (const float*)d_in[16];
  const float* ad1d = (const float*)d_in[17];
  const float* bd1  = (const float*)d_in[18];
  const float* wd2  = (const float*)d_in[19];
  const float* ad2s = (const float*)d_in[20];
  const float* ad2d = (const float*)d_in[21];
  const float* bd2  = (const float*)d_in[22];

  const int Nn = in_sizes[0] / 128;
  const int E  = in_sizes[1] / 2;

  // workspace layout (floats)
  float* ws    = (float*)d_ws;
  float* bufA  = ws;                               // GEMM output h   [N*128]
  float* bufB  = bufA  + (size_t)Nn * 128;         // agg             [N*128]
  float* bufC  = bufB  + (size_t)Nn * 128;         // h1 / d1         [N*128]
  float* als   = bufC  + (size_t)Nn * 128;         // [N*8]
  float* ald   = als   + (size_t)Nn * 8;           // [N*8]
  float* emax  = ald   + (size_t)Nn * 8;           // [N*8]
  float* denom = emax  + (size_t)Nn * 8;           // [N*8]
  float* zbuf  = denom + (size_t)Nn * 8;           // [N*32]
  _Float16* w16 = (_Float16*)(zbuf + (size_t)Nn * 32); // [<=128*128 halfs]

  float* out   = (float*)d_out;
  float* recon = out;                              // [N*128]
  float* mu    = out + (size_t)Nn * 128;           // [N*32]
  float* lv    = mu  + (size_t)Nn * 32;            // [N*32]

  // encoder1: 128 -> 8 heads x 16, ReLU -> h1 (bufC)
  run_gat(x, 128, w1, a1s, a1d, b1, 8, 16, bufC, 1,
          ei, E, Nn, bufA, bufB, als, ald, emax, denom, w16, stream);
  // encoder2_mu: 128 -> 1 x 32 -> mu (d_out)
  run_gat(bufC, 128, wmu, amus, amud, bmu, 1, 32, mu, 0,
          ei, E, Nn, bufA, bufB, als, ald, emax, denom, w16, stream);
  // encoder2_logvar: 128 -> 1 x 32 -> logvar (d_out)
  run_gat(bufC, 128, wlv, alvs, alvd, blv, 1, 32, lv, 0,
          ei, E, Nn, bufA, bufB, als, ald, emax, denom, w16, stream);
  // reparameterize
  z_kernel<<<ceil_div(Nn * 32, 256), 256, 0, stream>>>(eps, lv, mu, zbuf, Nn * 32);
  // decoder1: 32 -> 1 x 128, ReLU -> d1 (bufC, h1 no longer needed)
  run_gat(zbuf, 32, wd1, ad1s, ad1d, bd1, 1, 128, bufC, 1,
          ei, E, Nn, bufA, bufB, als, ald, emax, denom, w16, stream);
  // decoder2: 128 -> 8 heads x 16 -> recon (d_out)
  run_gat(bufC, 128, wd2, ad2s, ad2d, bd2, 8, 16, recon, 0,
          ei, E, Nn, bufA, bufB, als, ald, emax, denom, w16, stream);
}
